// GCN_38474317037931
// MI455X (gfx1250) — matrix operand order repaired and checked by reading, not compiled, not run on hardware
//
#include <hip/hip_runtime.h>

#define NN 100000
#define CC 128
#define EE 640000
#define GN_EPS 1e-5f
#define PITCH 136   // bf16 elements per LDS row (136*2=272 B, 16B aligned, bank-skewed)

typedef __attribute__((ext_vector_type(16))) __bf16 v16bf;
typedef __attribute__((ext_vector_type(8)))  __bf16 v8bf;
typedef __attribute__((ext_vector_type(8)))  float  v8f;

union Frag { v16bf v; v8bf h[2]; };

__device__ __forceinline__ __bf16 f2bf(float f) {
  unsigned u = __builtin_bit_cast(unsigned, f);
  u += 0x7FFFu + ((u >> 16) & 1u);               // round-to-nearest-even
  unsigned short s = (unsigned short)(u >> 16);
  return __builtin_bit_cast(__bf16, s);
}
__device__ __forceinline__ float bf2f(__bf16 b) {
  unsigned short s = __builtin_bit_cast(unsigned short, b);
  return __builtin_bit_cast(float, ((unsigned)s) << 16);
}

// gfx1250 async DMA: 16B global -> LDS per lane, tracked by ASYNCcnt.
__device__ __forceinline__ void async_copy16(unsigned ldsAddr, const void* gaddr) {
  asm volatile("global_load_async_to_lds_b128 %0, %1, off"
               :: "v"(ldsAddr), "v"(gaddr) : "memory");
}
__device__ __forceinline__ void wait_asynccnt0() {
  asm volatile("s_wait_asynccnt 0x0" ::: "memory");
}

// ---------------------------------------------------------------------------
// One-time per layer: split W (fp32, row-major [k][c]) into transposed bf16
// hi/lo images WT[c][k] so the GEMM can stage it with pure async byte copies.
// ---------------------------------------------------------------------------
__global__ void split_wT(const float* __restrict__ W,
                         __bf16* __restrict__ hiT, __bf16* __restrict__ loT) {
  int i = blockIdx.x * 256 + threadIdx.x;        // 64 blocks x 256 = 16384
  if (i >= CC * CC) return;
  int k = i >> 7, c = i & 127;
  float w = W[i];                                // W[k][c]
  __bf16 h = f2bf(w);
  hiT[(size_t)c * CC + k] = h;                   // transposed
  loT[(size_t)c * CC + k] = f2bf(w - bf2f(h));
}

// ---------------------------------------------------------------------------
// H = X @ W via split-bf16 WMMA (ah*bh + ah*bl + al*bh), f32 accumulate.
// Block = 256 threads (8 waves), computes 32 rows x 64 cols of H.
// W slab staged via global_load_async_to_lds_b128 (ASYNCcnt DMA).
// ---------------------------------------------------------------------------
__global__ __launch_bounds__(256)
void gemm_wmma(const float* __restrict__ X,
               const __bf16* __restrict__ WThi, const __bf16* __restrict__ WTlo,
               float* __restrict__ H) {
  __shared__ __bf16 sWThi[64][PITCH];   // W^T slab: [col][k]
  __shared__ __bf16 sWTlo[64][PITCH];
  __shared__ __bf16 sXhi[32][PITCH];    // X tile:  [row][k]
  __shared__ __bf16 sXlo[32][PITCH];

  const int t       = threadIdx.x;
  const int blk     = blockIdx.x;
  const int rowBlk  = blk >> 1;         // 0..3124
  const int slab    = blk & 1;          // which 64-col half of W
  const int rowBase = rowBlk * 32;
  const int colSlab = slab * 64;

  // Async DMA stage of the transposed W slab (hi+lo): 64 rows x 16 chunks x 16B.
  for (int i = t; i < 64 * 16; i += 256) {       // 4 iterations
    int c = i >> 4, ch = i & 15;
    size_t gOff = (size_t)(colSlab + c) * CC + ch * 8;   // bf16 elements
    async_copy16((unsigned)(size_t)(&sWThi[c][0]) + ch * 16, WThi + gOff);
    async_copy16((unsigned)(size_t)(&sWTlo[c][0]) + ch * 16, WTlo + gOff);
  }
  // Stage X tile (32 rows x 128 k), split hi/lo in VALU (changes per layer).
  for (int i = t; i < 32 * CC; i += 256) {
    int r = i >> 7, c = i & 127;
    float v = X[(size_t)(rowBase + r) * CC + c];
    __bf16 hi = f2bf(v);
    sXhi[r][c] = hi;
    sXlo[r][c] = f2bf(v - bf2f(hi));
  }
  wait_asynccnt0();
  __syncthreads();

  const int lane    = t & 31;
  const int wave    = t >> 5;
  const int rt      = wave & 1;         // row tile within block (0..1)
  const int ct      = wave >> 1;        // col tile within slab  (0..3)
  const int mn      = lane & 15;        // A row / B col / D col index
  const bool hiHalf = lane >= 16;
  const int rowLoc  = rt * 16 + mn;     // local X row for A fragment
  const int colLoc  = ct * 16 + mn;     // local W^T row for B fragment

  v8f acc = {};

  #pragma unroll
  for (int kc = 0; kc < 4; ++kc) {
    const int K0 = kc * 32;
    // A fragment (16-bit A 16x32 layout): lanes<16 hold K=[0..8)+[16..24),
    // lanes>=16 hold K=[8..16)+[24..32) of the chunk.
    const int ka0 = K0 + (hiHalf ? 8 : 0);
    const int ka1 = K0 + (hiHalf ? 24 : 16);
    Frag ah, al;
    ah.h[0] = *(const v8bf*)&sXhi[rowLoc][ka0];
    ah.h[1] = *(const v8bf*)&sXhi[rowLoc][ka1];
    al.h[0] = *(const v8bf*)&sXlo[rowLoc][ka0];
    al.h[1] = *(const v8bf*)&sXlo[rowLoc][ka1];
    // B fragment: lane holds one column; lanes<16 K=[0..8)+[16..24), lanes>=16 K=[8..16)+[24..32).
    const int kb = K0 + (hiHalf ? 8 : 0);
    Frag bh, bl;
    bh.h[0] = *(const v8bf*)&sWThi[colLoc][kb];
    bh.h[1] = *(const v8bf*)&sWThi[colLoc][kb + 16];
    bl.h[0] = *(const v8bf*)&sWTlo[colLoc][kb];
    bl.h[1] = *(const v8bf*)&sWTlo[colLoc][kb + 16];

    acc = __builtin_amdgcn_wmma_f32_16x16x32_bf16(false, ah.v, false, bh.v,
                                                  (short)0, acc, false, false);
    acc = __builtin_amdgcn_wmma_f32_16x16x32_bf16(false, ah.v, false, bl.v,
                                                  (short)0, acc, false, false);
    acc = __builtin_amdgcn_wmma_f32_16x16x32_bf16(false, al.v, false, bh.v,
                                                  (short)0, acc, false, false);
  }

  // D layout: VGPR r -> row (hiHalf ? r+8 : r), col = mn.
  const int col = colSlab + ct * 16 + mn;
  #pragma unroll
  for (int r = 0; r < 8; ++r) {
    int row = rowBase + rt * 16 + (hiHalf ? r + 8 : r);
    H[(size_t)row * CC + col] = acc[r];
  }
}

// ---------------------------------------------------------------------------
// Degree / normalization helpers
// ---------------------------------------------------------------------------
__global__ void fill_ones(float* __restrict__ deg) {
  int i = blockIdx.x * 256 + threadIdx.x;
  if (i < NN) deg[i] = 1.0f;               // self-loop contribution
}
__global__ void deg_accum(const int* __restrict__ dst, float* __restrict__ deg) {
  int e = blockIdx.x * 256 + threadIdx.x;
  if (e < EE) atomicAdd(&deg[dst[e]], 1.0f);
}
__global__ void make_dinv(const float* __restrict__ deg, float* __restrict__ dinv) {
  int i = blockIdx.x * 256 + threadIdx.x;
  if (i < NN) { float d = deg[i]; dinv[i] = d > 0.f ? rsqrtf(d) : 0.f; }
}

// out[i,c] = bias[c] + h[i,c]*dinv[i]^2   (bias + self-loop term, full init)
__global__ void init_out(float* __restrict__ out, const float* __restrict__ h,
                         const float* __restrict__ bias, const float* __restrict__ dinv) {
  size_t idx = (size_t)blockIdx.x * 256 + threadIdx.x;
  int c = (int)(idx & 127);
  size_t i = idx >> 7;
  float di = dinv[i];
  out[idx] = bias[c] + h[idx] * di * di;
}

// 32 lanes per edge, 4 channels (one float4) per lane.
__global__ void scatter_edges(const int* __restrict__ src, const int* __restrict__ dst,
                              const float* __restrict__ h, const float* __restrict__ dinv,
                              float* __restrict__ out) {
  size_t gid = (size_t)blockIdx.x * 256 + threadIdx.x;
  int e = (int)(gid >> 5);
  if (e >= EE) return;
  int q = (int)(gid & 31);
  int s = src[e], d = dst[e];
  float norm = dinv[s] * dinv[d];
  float4 v = ((const float4*)(h + (size_t)s * CC))[q];
  float* o = out + (size_t)d * CC + q * 4;
  atomicAdd(o + 0, v.x * norm);
  atomicAdd(o + 1, v.y * norm);
  atomicAdd(o + 2, v.z * norm);
  atomicAdd(o + 3, v.w * norm);
}

// ---------------------------------------------------------------------------
// GraphNorm + ReLU (fused): stats -> finalize coefs -> apply
// ---------------------------------------------------------------------------
__global__ void zero_stats(float* __restrict__ s) { s[threadIdx.x] = 0.f; }

__global__ void gn_stats(const float* __restrict__ x, float* __restrict__ stats) {
  int c = threadIdx.x & 127;
  int half = threadIdx.x >> 7;
  float s = 0.f, q = 0.f;
  for (int row = blockIdx.x * 2 + half; row < NN; row += gridDim.x * 2) {
    float v = x[(size_t)row * CC + c];
    s += v; q += v * v;
  }
  atomicAdd(&stats[c], s);
  atomicAdd(&stats[128 + c], q);
}

__global__ void gn_finalize(const float* __restrict__ stats,
                            const float* __restrict__ w, const float* __restrict__ b,
                            const float* __restrict__ ms, float* __restrict__ coef) {
  int c = threadIdx.x;                       // 128 threads
  const float invN = 1.0f / (float)NN;
  float mean = stats[c] * invN;
  float ex2  = stats[128 + c] * invN;
  float m2   = ms[c] * mean;                 // ms * mean
  float var  = ex2 - 2.f * m2 * mean + m2 * m2;   // E[(x - ms*mean)^2]
  float inv  = rsqrtf(var + GN_EPS);
  coef[c]       = w[c] * inv;                // scale
  coef[128 + c] = b[c] - w[c] * inv * m2;    // shift
}

__global__ void gn_apply(float* __restrict__ x, const float* __restrict__ coef) {
  size_t idx = (size_t)blockIdx.x * 256 + threadIdx.x;
  int c = (int)(idx & 127);
  float v = coef[c] * x[idx] + coef[128 + c];
  x[idx] = v > 0.f ? v : 0.f;
}

// ---------------------------------------------------------------------------
extern "C" void kernel_launch(void* const* d_in, const int* in_sizes, int n_in,
                              void* d_out, int out_size, void* d_ws, size_t ws_size,
                              hipStream_t stream) {
  const float* x   = (const float*)d_in[0];
  const int*   ei  = (const int*)d_in[1];
  const int*   src = ei;
  const int*   dst = ei + EE;
  const float* W1 = (const float*)d_in[2];  const float* b1 = (const float*)d_in[3];
  const float* g1w = (const float*)d_in[4]; const float* g1b = (const float*)d_in[5];
  const float* g1m = (const float*)d_in[6];
  const float* W2 = (const float*)d_in[7];  const float* b2 = (const float*)d_in[8];
  const float* g2w = (const float*)d_in[9]; const float* g2b = (const float*)d_in[10];
  const float* g2m = (const float*)d_in[11];
  const float* W3 = (const float*)d_in[12]; const float* b3 = (const float*)d_in[13];

  float* out = (float*)d_out;
  float* ws  = (float*)d_ws;
  const size_t NC = (size_t)NN * CC;
  float* bufA  = ws;              // N x C
  float* bufB  = ws + NC;         // N x C
  float* deg   = ws + 2 * NC;     // N
  float* dinv  = deg + NN;        // N
  float* stats = dinv + NN;       // 256
  float* coef  = stats + 256;     // 256
  // bf16 split-W images (transposed), 16384 bf16 each = 32 KB; 6 images total.
  __bf16* wsplit = (__bf16*)(coef + 256);
  __bf16* W1hi = wsplit;              __bf16* W1lo = wsplit + 16384;
  __bf16* W2hi = wsplit + 2 * 16384;  __bf16* W2lo = wsplit + 3 * 16384;
  __bf16* W3hi = wsplit + 4 * 16384;  __bf16* W3lo = wsplit + 5 * 16384;

  const int gbGemm = (NN / 32) * 2;          // 6250
  const int gbNC   = (int)(NC / 256);        // 50000
  const int gbScat = (EE * 32) / 256;        // 80000
  const int gbN    = (NN + 255) / 256;
  const int gbE    = (EE + 255) / 256;

  // Degrees (shared across all three layers) + one-time W splits
  fill_ones<<<gbN, 256, 0, stream>>>(deg);
  deg_accum<<<gbE, 256, 0, stream>>>(dst, deg);
  make_dinv<<<gbN, 256, 0, stream>>>(deg, dinv);
  split_wT<<<64, 256, 0, stream>>>(W1, W1hi, W1lo);
  split_wT<<<64, 256, 0, stream>>>(W2, W2hi, W2lo);
  split_wT<<<64, 256, 0, stream>>>(W3, W3hi, W3lo);

  // ----- Layer 1 (h scratch = d_out) -----
  gemm_wmma<<<gbGemm, 256, 0, stream>>>(x, W1hi, W1lo, out);
  init_out<<<gbNC, 256, 0, stream>>>(bufA, out, b1, dinv);
  scatter_edges<<<gbScat, 256, 0, stream>>>(src, dst, out, dinv, bufA);
  zero_stats<<<1, 256, 0, stream>>>(stats);
  gn_stats<<<512, 256, 0, stream>>>(bufA, stats);
  gn_finalize<<<1, 128, 0, stream>>>(stats, g1w, g1b, g1m, coef);
  gn_apply<<<gbNC, 256, 0, stream>>>(bufA, coef);

  // ----- Layer 2 (h scratch = d_out) -----
  gemm_wmma<<<gbGemm, 256, 0, stream>>>(bufA, W2hi, W2lo, out);
  init_out<<<gbNC, 256, 0, stream>>>(bufB, out, b2, dinv);
  scatter_edges<<<gbScat, 256, 0, stream>>>(src, dst, out, dinv, bufB);
  zero_stats<<<1, 256, 0, stream>>>(stats);
  gn_stats<<<512, 256, 0, stream>>>(bufB, stats);
  gn_finalize<<<1, 128, 0, stream>>>(stats, g2w, g2b, g2m, coef);
  gn_apply<<<gbNC, 256, 0, stream>>>(bufB, coef);

  // ----- Layer 3 (h scratch = bufA, final result -> d_out) -----
  gemm_wmma<<<gbGemm, 256, 0, stream>>>(bufB, W3hi, W3lo, bufA);
  init_out<<<gbNC, 256, 0, stream>>>(out, bufA, b3, dinv);
  scatter_edges<<<gbScat, 256, 0, stream>>>(src, dst, bufA, dinv, out);
}
